// GraphEncoder_47330539602050
// MI455X (gfx1250) — matrix operand order repaired and checked by reading, not compiled, not run on hardware
//
#include <hip/hip_runtime.h>

typedef __attribute__((ext_vector_type(16))) _Float16 v16h;
typedef __attribute__((ext_vector_type(8)))  float    v8f;

#define N_NODES 20000
#define N_EDGES 640000
#define D 128
#define LDS_STRIDE 136   // halves per row; 272B row stride dodges 64-bank conflicts

// ---------------------------------------------------------------- zero fill
__global__ __launch_bounds__(256) void zero_f32(float* __restrict__ p, int n4) {
  int i = blockIdx.x * blockDim.x + threadIdx.x;
  if (i < n4) ((float4*)p)[i] = make_float4(0.f, 0.f, 0.f, 0.f);
}

// -------------------------------------------- weight pre-swizzle (per launch)
// f32 row-major [K=128][N=128]  ->  f16 WMMA B-fragment order:
//   halves[ ((nt*4 + kt)*32 + lane)*16 + i ] = w[(kt*32 + (lane>>4)*16 + i)*128
//                                                + nt*16 + (lane&15)]
// so the hot loop loads each B fragment as one contiguous 32-byte vector.
__global__ __launch_bounds__(256) void swizzle_w(const float* __restrict__ w,
                                                 _Float16* __restrict__ o) {
  int t = blockIdx.x * 256 + threadIdx.x;   // 0..1023 : (frag, lane)
  if (t >= 32 * 32) return;
  int frag = t >> 5;
  int lane = t & 31;
  int nt = frag >> 2, kt = frag & 3;
  int n = lane & 15, kh = lane >> 4;
  const float* wp = w + (size_t)(kt * 32 + kh * 16) * D + nt * 16 + n;
  _Float16* op = o + (size_t)t * 16;
  #pragma unroll
  for (int i = 0; i < 16; ++i) op[i] = (_Float16)wp[i * D];
}

// ------------------------------------------------- edge scatter-add (GIN agg)
// one wave32 per edge; lane l handles dims [4l, 4l+3]
__global__ __launch_bounds__(256) void gin_aggregate(
    const float* __restrict__ z, const int* __restrict__ src,
    const int* __restrict__ dst, float* __restrict__ agg) {
  int t = blockIdx.x * blockDim.x + threadIdx.x;
  int e = t >> 5;
  int lane = t & 31;
  if (e >= N_EDGES) return;
  int s = src[e];
  int d = dst[e];
  float4 v = ((const float4*)(z + (size_t)s * D))[lane];
  float* ap = agg + (size_t)d * D + lane * 4;
  unsafeAtomicAdd(ap + 0, v.x);   // global_atomic_add_f32, L2-resident
  unsafeAtomicAdd(ap + 1, v.y);
  unsafeAtomicAdd(ap + 2, v.z);
  unsafeAtomicAdd(ap + 3, v.w);
}

// -------------------------------------------- fused  relu(relu((z+agg)W1+b1)W2+b2)
// 256 threads = 8 waves; block tile = 128 rows; wave tile = 16 rows x 128 cols.
__global__ __launch_bounds__(256) void gin_mlp(
    const float* __restrict__ z, const float* __restrict__ agg,
    const _Float16* __restrict__ w1s, const float* __restrict__ b1,
    const _Float16* __restrict__ w2s, const float* __restrict__ b2,
    float* __restrict__ out) {
  __shared__ _Float16 hS[128 * LDS_STRIDE];   // 34,816 B

  const int tid  = threadIdx.x;
  const int row0 = blockIdx.x * 128;

  // ---- Phase 1: h = z + agg -> f16 in LDS (reads clamped, stores guarded later)
  for (int idx = tid; idx < 128 * (D / 4); idx += 256) {
    int r  = idx >> 5;        // row in tile
    int c4 = idx & 31;        // float4 column
    int gr = row0 + r;
    if (gr >= N_NODES) gr = N_NODES - 1;
    float4 zv = ((const float4*)(z   + (size_t)gr * D))[c4];
    float4 av = ((const float4*)(agg + (size_t)gr * D))[c4];
    _Float16* p = &hS[r * LDS_STRIDE + c4 * 4];
    p[0] = (_Float16)(zv.x + av.x);
    p[1] = (_Float16)(zv.y + av.y);
    p[2] = (_Float16)(zv.z + av.z);
    p[3] = (_Float16)(zv.w + av.w);
  }
  __syncthreads();

  const int lane    = tid & 31;
  const int wv      = tid >> 5;
  const int m       = lane & 15;   // A row in tile == B/D column in tile
  const int kh      = lane >> 4;   // lane half-group
  const int waveRow = wv * 16;

  const v16h* W1 = (const v16h*)w1s;   // fragment-ordered, 32B per (frag,lane)
  const v16h* W2 = (const v16h*)w2s;

  // ================= GEMM1: hid = relu(h @ W1 + b1) =================
  v16h a0, a1, a2, a3;
  {
    const _Float16* base = &hS[(waveRow + m) * LDS_STRIDE];
    #pragma unroll
    for (int i = 0; i < 16; ++i) {
      const int k = kh * 16 + i;  // contiguous k order, matches swizzled B
      a0[i] = base[k];
      a1[i] = base[32 + k];
      a2[i] = base[64 + k];
      a3[i] = base[96 + k];
    }
  }
  // all A data for this wave is now in registers -> safe to overwrite hS rows
  #pragma unroll
  for (int nt = 0; nt < 8; ++nt) {
    const int col  = nt * 16 + m;
    const float bs = b1[col];
    v8f acc;
    #pragma unroll
    for (int r = 0; r < 8; ++r) acc[r] = bs;   // bias folded into C
    #pragma unroll
    for (int kt = 0; kt < 4; ++kt) {
      const v16h b = W1[(nt * 4 + kt) * 32 + lane];
      const v16h a = (kt == 0) ? a0 : (kt == 1) ? a1 : (kt == 2) ? a2 : a3;
      acc = __builtin_amdgcn_wmma_f32_16x16x32_f16(false, a, false, b,
                                                   (short)0, acc, false, false);
    }
    #pragma unroll
    for (int r = 0; r < 8; ++r) {              // ISA C/D layout: row = 8*kh + r
      float v = acc[r] > 0.f ? acc[r] : 0.f;
      hS[(waveRow + kh * 8 + r) * LDS_STRIDE + col] = (_Float16)v;
    }
  }
  __syncthreads();

  // ================= GEMM2: out = relu(hid @ W2 + b2) =================
  {
    const _Float16* base = &hS[(waveRow + m) * LDS_STRIDE];
    #pragma unroll
    for (int i = 0; i < 16; ++i) {
      const int k = kh * 16 + i;
      a0[i] = base[k];
      a1[i] = base[32 + k];
      a2[i] = base[64 + k];
      a3[i] = base[96 + k];
    }
  }
  #pragma unroll
  for (int nt = 0; nt < 8; ++nt) {
    const int col  = nt * 16 + m;
    const float bs = b2[col];
    v8f acc;
    #pragma unroll
    for (int r = 0; r < 8; ++r) acc[r] = bs;
    #pragma unroll
    for (int kt = 0; kt < 4; ++kt) {
      const v16h b = W2[(nt * 4 + kt) * 32 + lane];
      const v16h a = (kt == 0) ? a0 : (kt == 1) ? a1 : (kt == 2) ? a2 : a3;
      acc = __builtin_amdgcn_wmma_f32_16x16x32_f16(false, a, false, b,
                                                   (short)0, acc, false, false);
    }
    #pragma unroll
    for (int r = 0; r < 8; ++r) {
      float v = acc[r] > 0.f ? acc[r] : 0.f;
      int gr = row0 + waveRow + kh * 8 + r;
      if (gr < N_NODES) out[(size_t)gr * D + col] = v;
    }
  }
}

// ---------------------------------------------------------------- launcher
extern "C" void kernel_launch(void* const* d_in, const int* in_sizes, int n_in,
                              void* d_out, int out_size, void* d_ws, size_t ws_size,
                              hipStream_t stream) {
  const float* x  = (const float*)d_in[0];
  const int*   ei = (const int*)d_in[1];   // JAX default x64-off -> int32
  const int*  src = ei;
  const int*  dst = ei + N_EDGES;

  const float* W[3][4];
  for (int l = 0; l < 3; ++l)
    for (int j = 0; j < 4; ++j)
      W[l][j] = (const float*)d_in[2 + 4 * l + j];

  float* out = (float*)d_out;

  // ws layout: agg (10.24 MB, 32B-aligned) then 6 swizzled f16 weights (32KB each)
  float* agg = (float*)d_ws;
  _Float16* wsw = (_Float16*)((char*)d_ws + (size_t)N_NODES * D * sizeof(float));
  const size_t WHALF = (size_t)D * D;      // 16384 halves per weight

  // one-time (per launch) weight swizzle+convert: f32 row-major -> f16 B-fragments
  for (int l = 0; l < 3; ++l) {
    swizzle_w<<<4, 256, 0, stream>>>(W[l][0], wsw + (2 * l + 0) * WHALF);
    swizzle_w<<<4, 256, 0, stream>>>(W[l][2], wsw + (2 * l + 1) * WHALF);
  }

  const int aggN4 = N_NODES * D / 4;
  dim3 zgrid((aggN4 + 255) / 256);
  dim3 agrid((N_EDGES * 32) / 256);        // 80000 blocks, one wave per edge
  dim3 mgrid((N_NODES + 127) / 128);       // 157 blocks

  const float* zin = x;
  for (int l = 0; l < 3; ++l) {
    zero_f32<<<zgrid, 256, 0, stream>>>(agg, aggN4);
    gin_aggregate<<<agrid, 256, 0, stream>>>(zin, src, dst, agg);
    gin_mlp<<<mgrid, 256, 0, stream>>>(zin, agg,
                                       wsw + (2 * l + 0) * WHALF, W[l][1],
                                       wsw + (2 * l + 1) * WHALF, W[l][3], out);
    zin = out;                             // in-place chaining through d_out
  }
}
